// UNet_43516608643454
// MI455X (gfx1250) — matrix operand order repaired and checked by reading, not compiled, not run on hardware
//
#include <hip/hip_runtime.h>

// ---------------------------------------------------------------------------
// SelectionConv Graph-UNet for MI455X (gfx1250, wave32).
// All channel GEMMs run through v_wmma_f32_16x16x32_f16 (one wave = one 16x16
// output tile, K stepped by 32). Activations stored f16; BN+bias+ReLU+residual
// fused into the WMMA epilogue. Gathers exploit dst*9+sel == edge_id (no
// atomics). Deterministic bump allocation over d_ws; every launch on `stream`.
//
// Input flattening assumed in setup_inputs() insertion order:
//   0        : x                      [65536*3] f32
//   1..6     : edge_indexes[0..5]     [2, n*9]  i32 (row0=src, row1=dst)
//   7..12    : selections_list[0..5]  [n*9]     i32
//   13..18   : interps_list[0..5]     [n*9]     f32
//   19..23   : clusters[0..4]         (unused; structure is arange//4)
//   24..27   : start_w, start_b, final_w, final_b
//   28+b*20  : block b of {enc1..enc5, dec5..dec1}:
//              w1,b1,w2,b2,w3,b3,wr,br, bn1{g,b,m,v}, bn2{g,b,m,v}, bnr{g,b,m,v}
// ---------------------------------------------------------------------------

typedef _Float16 half_t;
typedef __attribute__((ext_vector_type(16))) _Float16 v16h;
typedef __attribute__((ext_vector_type(8)))  float    v8f;

#define DSEL 9
#define BN_EPS 1e-5f

#define GSL(i, total)                                                          \
  for (long long i = (long long)blockIdx.x * blockDim.x + threadIdx.x;         \
       i < (total); i += (long long)gridDim.x * blockDim.x)

// ------------------------------ elementwise --------------------------------

__global__ void k_zero_h(half_t* __restrict__ p, long long n) {
  GSL(i, n) p[i] = (half_t)0.f;
}

__global__ void k_prep_ss(const float* __restrict__ gamma,
                          const float* __restrict__ beta,
                          const float* __restrict__ mean,
                          const float* __restrict__ var,
                          const float* __restrict__ bias,
                          float* __restrict__ scale, float* __restrict__ shift,
                          int C, int has_bn) {
  int c = blockIdx.x * blockDim.x + threadIdx.x;
  if (c >= C) return;
  if (has_bn) {
    float s = gamma[c] * rsqrtf(var[c] + BN_EPS);
    scale[c] = s;
    shift[c] = beta[c] - mean[c] * s + bias[c] * s;
  } else {
    scale[c] = 1.0f;
    shift[c] = bias[c];
  }
}

// w: [K, M] f32 row-major  ->  wt: [M, Kpad] f16 (transposed, zero-padded K)
__global__ void k_trans_w(const float* __restrict__ w, half_t* __restrict__ wt,
                          int K, int M, int Kpad) {
  GSL(i, (long long)M * Kpad) {
    int m = (int)(i / Kpad);
    int k = (int)(i % Kpad);
    wt[i] = (k < K) ? (half_t)w[(long long)k * M + m] : (half_t)0.f;
  }
}

// gather from f32 source (network input x)
__global__ void k_gather_f32(const float* __restrict__ x,
                             const int* __restrict__ src,
                             const int* __restrict__ dst,
                             const int* __restrict__ sel,
                             const float* __restrict__ itp,
                             half_t* __restrict__ agg, long long total,
                             int Cin, int Kpad) {
  GSL(i, total) {
    long long e = i / Cin;
    int c = (int)(i % Cin);
    int s = src[e], d = dst[e], q = sel[e];
    float v = x[(long long)s * Cin + c] * itp[e];
    agg[(long long)d * Kpad + q * Cin + c] = (half_t)v;
  }
}

// gather from f16 activations
__global__ void k_gather_f16(const half_t* __restrict__ x,
                             const int* __restrict__ src,
                             const int* __restrict__ dst,
                             const int* __restrict__ sel,
                             const float* __restrict__ itp,
                             half_t* __restrict__ agg, long long total,
                             int Cin, int Kpad) {
  GSL(i, total) {
    long long e = i / Cin;
    int c = (int)(i % Cin);
    int s = src[e], d = dst[e], q = sel[e];
    float v = (float)x[(long long)s * Cin + c] * itp[e];
    agg[(long long)d * Kpad + q * Cin + c] = (half_t)v;
  }
}

// max-pool over 4 consecutive rows (clusters = arange // 4)
__global__ void k_pool4(const half_t* __restrict__ in, half_t* __restrict__ out,
                        long long total, int C) {
  GSL(i, total) {
    long long r = i / C;
    int c = (int)(i % C);
    long long b = r * 4 * C + c;
    float m = (float)in[b];
    m = fmaxf(m, (float)in[b + C]);
    m = fmaxf(m, (float)in[b + 2 * C]);
    m = fmaxf(m, (float)in[b + 3 * C]);
    out[i] = (half_t)m;
  }
}

// out[i] = concat(up[i/4], skip[i])
__global__ void k_unpool_concat(const half_t* __restrict__ up,
                                const half_t* __restrict__ sk,
                                half_t* __restrict__ out, long long total,
                                int Cup, int Csk) {
  int Ct = Cup + Csk;
  GSL(i, total) {
    long long r = i / Ct;
    int c = (int)(i % Ct);
    out[i] = (c < Cup) ? up[(r >> 2) * Cup + c] : sk[r * Csk + (c - Cup)];
  }
}

// ------------------------------ WMMA GEMM ----------------------------------
// Y[n,M] = epilogue(A[n,Kpad] @ Wt^T), Wt stored [M,Kpad] f16.
// One wave per 16x16 tile; n is always a multiple of 16; Kpad multiple of 32.
// epilogue: y = acc*scale[col] + shift[col]; (+res) ; (ReLU) ; store f16/f32.
__global__ __launch_bounds__(256) void k_wmma_gemm(
    const half_t* __restrict__ A, const half_t* __restrict__ Wt,
    const float* __restrict__ scale, const float* __restrict__ shift,
    const float* __restrict__ resbuf, half_t* __restrict__ out16,
    float* __restrict__ out32, int n, int M, int Kpad, int relu) {
  const int lane = threadIdx.x & 31;
  const int wave = threadIdx.x >> 5;
  const int tiles_m = (M + 15) >> 4;
  const int tiles_n = n >> 4;
  int tile = blockIdx.x * 8 + wave;
  if (tile >= tiles_m * tiles_n) return;   // whole wave exits; EXEC all-ones below
  const int trow = tile / tiles_m;
  const int tcol = tile - trow * tiles_m;
  const int hi = (lane >> 4) & 1;          // wave32 half select
  const int l15 = lane & 15;
  const int rowA = (trow << 4) + l15;      // A: lanes 0-15 and 16-31 share rows
  const int col = (tcol << 4) + l15;       // B/C/D column
  const int colB = (col < M) ? col : 0;    // clamp; junk columns discarded at store
  const half_t* __restrict__ arow = A + (long long)rowA * Kpad;
  const half_t* __restrict__ brow = Wt + (long long)colB * Kpad;

  v8f acc = {};
  for (int k0 = 0; k0 < Kpad; k0 += 32) {
    v16h a, b;
#pragma unroll
    for (int v = 0; v < 8; ++v) {
      // A 16x32 f16 layout: VGPR v holds K = {2v,2v+1} (v<4) / {16+2(v-4),...},
      // +8 for lanes 16-31.
      int ka = ((v < 4) ? 2 * v : 16 + 2 * (v - 4)) + 8 * hi;
      a[2 * v] = arow[k0 + ka];
      a[2 * v + 1] = arow[k0 + ka + 1];
      // B 32x16 f16 layout: lane = column, same K order as A (+8 hi).
      int kb = ((v < 4) ? 2 * v : 16 + 2 * (v - 4)) + 8 * hi;
      b[2 * v] = brow[k0 + kb];
      b[2 * v + 1] = brow[k0 + kb + 1];
    }
    acc = __builtin_amdgcn_wmma_f32_16x16x32_f16(false, a, false, b, (short)0,
                                                 acc, false, false);
  }

  if (col < M) {
    float sc = scale[col], sh = shift[col];
#pragma unroll
    for (int r = 0; r < 8; ++r) {
      int row = (trow << 4) + r + 8 * hi;  // C/D layout: VGPR r -> M = r + 8*hi
      float y = acc[r] * sc + sh;
      long long o = (long long)row * M + col;
      if (resbuf) y += resbuf[o];
      if (relu) y = (y > 0.f) ? y : 0.f;
      if (out16) out16[o] = (half_t)y;
      if (out32) out32[o] = y;
    }
  }
}

// ------------------------------ host side ----------------------------------

static inline int nblk(long long total) {
  long long b = (total + 255) / 256;
  if (b < 1) b = 1;
  if (b > 1048576) b = 1048576;
  return (int)b;
}

extern "C" void kernel_launch(void* const* d_in, const int* in_sizes, int n_in,
                              void* d_out, int out_size, void* d_ws,
                              size_t ws_size, hipStream_t stream) {
  (void)in_sizes; (void)n_in; (void)out_size; (void)ws_size;
  static const int NS[6] = {65536, 16384, 4096, 1024, 256, 64};
  static const int encC[6] = {32, 64, 128, 256, 512, 512};

  auto F = [&](int i) { return (const float*)d_in[i]; };
  auto I = [&](int i) { return (const int*)d_in[i]; };

  size_t off = 0;
  auto alloc = [&](size_t bytes) -> char* {
    char* p = (char*)d_ws + off;
    off += (bytes + 255) & ~(size_t)255;
    return p;
  };

  // ---- persistent region: skip connections + dec ping/pong -------------
  half_t* encb[6];
  encb[0] = (half_t*)alloc((size_t)NS[0] * 32 * sizeof(half_t));       // h0
  for (int l = 1; l < 6; ++l)
    encb[l] = (half_t*)alloc((size_t)NS[l] * encC[l] * sizeof(half_t));
  half_t* ping = (half_t*)alloc((size_t)NS[0] * 32 * sizeof(half_t));
  half_t* pong = (half_t*)alloc((size_t)NS[0] * 32 * sizeof(half_t));
  const size_t t_base = off;

  // ---- launch helpers ---------------------------------------------------
  auto zero_h = [&](half_t* p, long long n) {
    k_zero_h<<<nblk(n), 256, 0, stream>>>(p, n);
  };
  auto gemm = [&](const half_t* A, const half_t* Wt, const float* sc,
                  const float* sh, const float* res, half_t* o16, float* o32,
                  int n, int M, int Kpad, int relu) {
    int tiles = (n >> 4) * ((M + 15) >> 4);
    k_wmma_gemm<<<(tiles + 7) / 8, 256, 0, stream>>>(A, Wt, sc, sh, res, o16,
                                                     o32, n, M, Kpad, relu);
  };
  auto prep_w = [&](const float* w, int K, int M, int Kpad) -> half_t* {
    half_t* wt = (half_t*)alloc((size_t)M * Kpad * sizeof(half_t));
    long long total = (long long)M * Kpad;
    k_trans_w<<<nblk(total), 256, 0, stream>>>(w, wt, K, M, Kpad);
    return wt;
  };
  auto prep_ss = [&](const float* g, const float* be, const float* me,
                     const float* va, const float* bias, int C,
                     int has_bn) -> float* {
    float* ss = (float*)alloc((size_t)2 * C * sizeof(float));
    k_prep_ss<<<(C + 255) / 256, 256, 0, stream>>>(g, be, me, va, bias, ss,
                                                   ss + C, C, has_bn);
    return ss;
  };
  auto gather16 = [&](const half_t* x, int lv, half_t* agg, int Cin, int Kpad) {
    int E = NS[lv] * DSEL;
    const int* ei = I(1 + lv);
    long long total = (long long)E * Cin;
    k_gather_f16<<<nblk(total), 256, 0, stream>>>(
        x, ei, ei + E, I(7 + lv), F(13 + lv), agg, total, Cin, Kpad);
  };

  // ---- residual block ---------------------------------------------------
  auto run_res = [&](int pbase, const half_t* xin, int lv, int cin, int cmid,
                     int cout, half_t* hout) {
    int n = NS[lv];
    // t1 = relu(bn1(xin @ w1 + b1))
    half_t* wt1 = prep_w(F(pbase + 0), cin, cmid, cin);
    float* ss1 = prep_ss(F(pbase + 8), F(pbase + 9), F(pbase + 10),
                         F(pbase + 11), F(pbase + 1), cmid, 1);
    half_t* t1 = (half_t*)alloc((size_t)n * cmid * sizeof(half_t));
    gemm(xin, wt1, ss1, ss1 + cmid, nullptr, t1, nullptr, n, cmid, cin, 1);
    // t2 = relu(bn2(sconv3(t1) @ w2 + b2))
    int K2 = DSEL * cmid;
    half_t* agg = (half_t*)alloc((size_t)n * K2 * sizeof(half_t));
    zero_h(agg, (long long)n * K2);
    gather16(t1, lv, agg, cmid, K2);
    half_t* wt2 = prep_w(F(pbase + 2), K2, cmid, K2);
    float* ss2 = prep_ss(F(pbase + 12), F(pbase + 13), F(pbase + 14),
                         F(pbase + 15), F(pbase + 3), cmid, 1);
    half_t* t2 = (half_t*)alloc((size_t)n * cmid * sizeof(half_t));
    gemm(agg, wt2, ss2, ss2 + cmid, nullptr, t2, nullptr, n, cmid, K2, 1);
    // res = bnr(xin @ wr + br)   (f32 scratch)
    half_t* wtr = prep_w(F(pbase + 6), cin, cout, cin);
    float* ssr = prep_ss(F(pbase + 16), F(pbase + 17), F(pbase + 18),
                         F(pbase + 19), F(pbase + 7), cout, 1);
    float* res = (float*)alloc((size_t)n * cout * sizeof(float));
    gemm(xin, wtr, ssr, ssr + cout, nullptr, nullptr, res, n, cout, cin, 0);
    // h = relu(t2 @ w3 + b3 + res)
    half_t* wt3 = prep_w(F(pbase + 4), cmid, cout, cmid);
    float* ss3 = prep_ss(nullptr, nullptr, nullptr, nullptr, F(pbase + 5),
                         cout, 0);
    gemm(t2, wt3, ss3, ss3 + cout, res, hout, nullptr, n, cout, cmid, 1);
  };

  // ---- start: h0 = sconv3(x) -------------------------------------------
  off = t_base;
  {
    int E = NS[0] * DSEL;
    half_t* agg0 = (half_t*)alloc((size_t)NS[0] * 32 * sizeof(half_t));
    zero_h(agg0, (long long)NS[0] * 32);  // pads cols 27..31
    long long total = (long long)E * 3;
    const int* ei = I(1);
    k_gather_f32<<<nblk(total), 256, 0, stream>>>(F(0), ei, ei + E, I(7),
                                                  F(13), agg0, total, 3, 32);
    half_t* wts = prep_w(F(24), 27, 32, 32);  // start_w [9,3,32] -> [27,32]
    float* sss = prep_ss(nullptr, nullptr, nullptr, nullptr, F(25), 32, 0);
    gemm(agg0, wts, sss, sss + 32, nullptr, encb[0], nullptr, NS[0], 32, 32, 0);
  }

  // ---- encoder ----------------------------------------------------------
  for (int l = 0; l < 5; ++l) {
    off = t_base;
    int nout = NS[l + 1], C = encC[l];
    half_t* p = (half_t*)alloc((size_t)nout * C * sizeof(half_t));
    long long total = (long long)nout * C;
    k_pool4<<<nblk(total), 256, 0, stream>>>(encb[l], p, total, C);
    run_res(28 + l * 20, p, l + 1, C, C, encC[l + 1], encb[l + 1]);
  }

  // ---- decoder ----------------------------------------------------------
  static const int decCm[5] = {256, 128, 64, 32, 32};
  half_t* hprev = encb[5];
  int Cprev = 512;
  half_t* slots[2] = {ping, pong};
  for (int j = 0; j < 5; ++j) {
    int lv = 4 - j;
    int Cskip = encC[lv];
    int cin = Cprev + Cskip;
    int n = NS[lv];
    off = t_base;
    half_t* xin = (half_t*)alloc((size_t)n * cin * sizeof(half_t));
    long long total = (long long)n * cin;
    k_unpool_concat<<<nblk(total), 256, 0, stream>>>(hprev, encb[lv], xin,
                                                     total, Cprev, Cskip);
    half_t* hout = slots[j & 1];
    run_res(28 + (5 + j) * 20, xin, lv, cin, decCm[j], decCm[j], hout);
    hprev = hout;
    Cprev = decCm[j];
  }

  // ---- final: sconv3 -> d_out [65536, 21] f32 ---------------------------
  off = t_base;
  {
    int K = DSEL * 32;  // 288
    half_t* agg = (half_t*)alloc((size_t)NS[0] * K * sizeof(half_t));
    zero_h(agg, (long long)NS[0] * K);
    gather16(hprev, 0, agg, 32, K);
    half_t* wtf = prep_w(F(26), K, 21, K);  // final_w [9,32,21] -> [288,21]
    float* ssf = prep_ss(nullptr, nullptr, nullptr, nullptr, F(27), 21, 0);
    gemm(agg, wtf, ssf, ssf + 21, nullptr, nullptr, (float*)d_out, NS[0], 21,
         K, 0);
  }
}
